// BertAttention_35588099014688
// MI455X (gfx1250) — compile-verified, matrix operand order checked by reading, not run on hardware
//
#include <hip/hip_runtime.h>
#include <hip/hip_bf16.h>

typedef __bf16 v16bf __attribute__((ext_vector_type(16)));
typedef float  v8f   __attribute__((ext_vector_type(8)));
typedef float  v2f   __attribute__((ext_vector_type(2)));

#define WMMA_BF16(a, b, c) \
  __builtin_amdgcn_wmma_f32_16x16x32_bf16(false, (a), false, (b), (short)0, (c), false, false)

#define V8F_ZERO {0.f,0.f,0.f,0.f,0.f,0.f,0.f,0.f}

// force GLOBAL (addrspace 1) vector memory ops instead of FLAT
#define AS1 __attribute__((address_space(1)))
template <typename T>
static __device__ __forceinline__ T ldg(const T* p) {
  return *(const AS1 T*)p;
}
template <typename T>
static __device__ __forceinline__ void stg(T* p, T v) {
  *(AS1 T*)p = v;
}

// ---------------------------------------------------------------------------
// Problem constants
// ---------------------------------------------------------------------------
#define BATCH 8
#define SEQ   1024
#define EMB   1024
#define NH    16
#define HD    64
#define ROWS  (BATCH * SEQ)      // 8192 GEMM rows

// workspace layout (bytes, all 256-aligned)
#define SZ_W      (1024ull * 1024ull * 2ull)              // one bf16 weight, 2 MB
#define OFF_WQT   (0ull)
#define OFF_WKT   (OFF_WQT + SZ_W)
#define OFF_WVT   (OFF_WKT + SZ_W)
#define OFF_WOT   (OFF_WVT + SZ_W)
#define OFF_EB    (OFF_WOT + SZ_W)                        // 2047*64 bf16 = 262016 B
#define OFF_QH    (OFF_EB + 512ull * 1024ull)
#define SZ_BHSD   ((size_t)BATCH * NH * SEQ * HD * 2ull)  // 16 MB
#define OFF_KH    (OFF_QH + SZ_BHSD)
#define OFF_VT    (OFF_KH + SZ_BHSD)
#define OFF_CTX   (OFF_VT + SZ_BHSD)                      // ctx bf16 [8192][1024]

// ---------------------------------------------------------------------------
// Kernel 0a: W[k][n] f32  ->  Wt[n][k] bf16   (makes WMMA B-frags contiguous)
// ---------------------------------------------------------------------------
__global__ void k_transpose_bf16(const float* __restrict__ W, __bf16* __restrict__ Wt) {
  int idx = blockIdx.x * 256 + threadIdx.x;   // 1M threads exactly
  int n = idx >> 10, k = idx & 1023;
  stg(Wt + idx, (__bf16)ldg(W + k * 1024 + n));   // Wt[n*1024+k]
}

// Kernel 0b: dist_emb f32 -> bf16
__global__ void k_cvt_emb(const float* __restrict__ E, __bf16* __restrict__ Eb) {
  int idx = blockIdx.x * 256 + threadIdx.x;
  if (idx < 2047 * 64) stg(Eb + idx, (__bf16)ldg(E + idx));
}

// ---------------------------------------------------------------------------
// Kernel 1: fused QKV projection.
//   Q -> Qh[b][h][s][d] bf16,  K -> Kh[b][h][s][d] bf16,  V -> Vt[b][h][d][s] bf16
// One workgroup (256 thr = 8 waves) per 16-row tile of X.
// ---------------------------------------------------------------------------
__global__ void __launch_bounds__(256)
k_qkv(const float* __restrict__ hs,
      const __bf16* __restrict__ Wqt, const __bf16* __restrict__ Wkt,
      const __bf16* __restrict__ Wvt,
      const float* __restrict__ bq, const float* __restrict__ bk,
      const float* __restrict__ bv,
      __bf16* __restrict__ Qh, __bf16* __restrict__ Kh, __bf16* __restrict__ Vt) {
  __shared__ unsigned int Xl[16 * 512];       // 16 rows x 1024 bf16 (packed pairs), 32 KB
  const int tid  = threadIdx.x;
  const int lane = tid & 31;
  const int w    = tid >> 5;
  const int l16  = lane & 15;
  const int hl   = lane >> 4;
  const int row0 = blockIdx.x * 16;

  // stage X tile: f32 -> bf16 packed into LDS
  for (int it = 0; it < 32; ++it) {
    int m  = tid + it * 256;                  // pair index in [0, 8192)
    int r  = m >> 9;                          // row 0..15
    int cp = m & 511;                         // pair within row
    v2f f = ldg((const v2f*)(hs + (size_t)(row0 + r) * EMB) + cp);
    __bf16 x = (__bf16)f.x, y = (__bf16)f.y;
    unsigned int pk = (unsigned int)__builtin_bit_cast(unsigned short, x) |
                      ((unsigned int)__builtin_bit_cast(unsigned short, y) << 16);
    Xl[r * 512 + cp] = pk;
  }
  __syncthreads();
  const __bf16* Xb = (const __bf16*)Xl;

  const __bf16* Wt3[3]   = {Wqt, Wkt, Wvt};
  const float*  bias3[3] = {bq, bk, bv};

  for (int mat = 0; mat < 3; ++mat) {
    const __bf16* wt = Wt3[mat];
    const float*  bi = bias3[mat];
    for (int g = 0; g < 2; ++g) {
      v8f acc0 = V8F_ZERO, acc1 = V8F_ZERO, acc2 = V8F_ZERO, acc3 = V8F_ZERO;
      const int n0 = (w + 8 * (4 * g + 0)) * 16;
      const int n1 = (w + 8 * (4 * g + 1)) * 16;
      const int n2 = (w + 8 * (4 * g + 2)) * 16;
      const int n3 = (w + 8 * (4 * g + 3)) * 16;
      for (int ks = 0; ks < 32; ++ks) {
        const int kk = ks * 32 + hl * 16;
        v16bf aX = *(const v16bf*)(Xb + l16 * 1024 + kk);
        if (ks + 1 < 32)
          __builtin_prefetch(wt + (size_t)(n0 + l16) * 1024 + kk + 32, 0, 3);
        v16bf b0 = ldg((const v16bf*)(wt + (size_t)(n0 + l16) * 1024 + kk));
        acc0 = WMMA_BF16(aX, b0, acc0);
        v16bf b1 = ldg((const v16bf*)(wt + (size_t)(n1 + l16) * 1024 + kk));
        acc1 = WMMA_BF16(aX, b1, acc1);
        v16bf b2 = ldg((const v16bf*)(wt + (size_t)(n2 + l16) * 1024 + kk));
        acc2 = WMMA_BF16(aX, b2, acc2);
        v16bf b3 = ldg((const v16bf*)(wt + (size_t)(n3 + l16) * 1024 + kk));
        acc3 = WMMA_BF16(aX, b3, acc3);
      }
      v8f*  accs[4] = {&acc0, &acc1, &acc2, &acc3};
      int   ncol[4] = {n0, n1, n2, n3};
#pragma unroll
      for (int q = 0; q < 4; ++q) {
        const int n = ncol[q] + l16;
        const int hh = n >> 6, dd = n & 63;
        const float bias = ldg(bi + n);
#pragma unroll
        for (int v = 0; v < 8; ++v) {
          const int r = row0 + v + 8 * hl;
          const int s = r & (SEQ - 1);
          const int b = r >> 10;
          float val = (*accs[q])[v] + bias;
          if (mat == 0)
            stg(Qh + (size_t)(((b * NH + hh) * SEQ + s)) * HD + dd, (__bf16)val);
          else if (mat == 1)
            stg(Kh + (size_t)(((b * NH + hh) * SEQ + s)) * HD + dd, (__bf16)val);
          else
            stg(Vt + (size_t)(((b * NH + hh) * HD + dd)) * SEQ + s, (__bf16)val);
        }
      }
    }
  }
}

// ---------------------------------------------------------------------------
// Kernel 2: attention for one (b, h, 64-query-row block).
//   scores = (Q K^T + relq + relk) / 8 ; softmax ; ctx = P V
// LDS: 64x1024 f32 scores (256 KB) + 8 waves x (16x32 Gq + 16x32 Gk) f32 (32 KB)
// ---------------------------------------------------------------------------
__global__ void __launch_bounds__(256)
k_attn(const __bf16* __restrict__ Qh, const __bf16* __restrict__ Kh,
       const __bf16* __restrict__ Vt, const __bf16* __restrict__ Eb,
       __bf16* __restrict__ Ctx) {
  extern __shared__ char smem[];
  float* sc    = (float*)smem;                          // 64 * 1024 f32
  float* stage = (float*)(smem + 64 * 1024 * 4);        // 8 waves * 1024 f32

  const int tid  = threadIdx.x;
  const int lane = tid & 31;
  const int w    = tid >> 5;
  const int l16  = lane & 15;
  const int hl   = lane >> 4;

  const int bid = blockIdx.x;
  const int qb  = bid & 15;            // query block (64 rows each)
  const int h   = (bid >> 4) & 15;
  const int b   = bid >> 8;

  const __bf16* Qp = Qh + (size_t)(b * NH + h) * SEQ * HD;
  const __bf16* Kp = Kh + (size_t)(b * NH + h) * SEQ * HD;
  const __bf16* Vp = Vt + (size_t)(b * NH + h) * HD * SEQ;

  float* gq = stage + w * 1024;
  float* gk = gq + 512;

  // ---- pass A: score tiles ------------------------------------------------
  for (int c8 = 0; c8 < 8; ++c8) {
    const int ct = c8 * 8 + w;
    const int r0 = ct * 16;
    for (int rt = 0; rt < 4; ++rt) {
      const int q0 = qb * 64 + rt * 16;
      const int d0 = q0 - r0 + 1023;   // Toeplitz center index into dist_emb
      v8f s = V8F_ZERO, g0 = V8F_ZERO, g1 = V8F_ZERO, g2 = V8F_ZERO, g3 = V8F_ZERO;
#pragma unroll
      for (int kc = 0; kc < 2; ++kc) {
        const int ko  = kc * 32 + hl * 16;
        v16bf aQ = ldg((const v16bf*)(Qp + (size_t)(q0 + l16) * HD + ko));
        v16bf aK = ldg((const v16bf*)(Kp + (size_t)(r0 + l16) * HD + ko));
        int elo = d0 - 15 + l16;
        int ehi = elo + 16;
        ehi = (ehi > 2046) ? 2046 : ehi;   // pad column, value unused
        v16bf bElo = ldg((const v16bf*)(Eb + (size_t)elo * HD + ko));
        v16bf bEhi = ldg((const v16bf*)(Eb + (size_t)ehi * HD + ko));
        s  = WMMA_BF16(aQ, aK,   s);   // Q K^T (K-frag doubles as B of K^T)
        g0 = WMMA_BF16(aQ, bElo, g0);  // Gq cols 0..15
        g1 = WMMA_BF16(aQ, bEhi, g1);  // Gq cols 16..31
        g2 = WMMA_BF16(aK, bElo, g2);  // Gk cols 0..15
        g3 = WMMA_BF16(aK, bEhi, g3);  // Gk cols 16..31
      }
      // stage Gq/Gk (wave-private, same-wave LDS ordering -> no barrier)
#pragma unroll
      for (int v = 0; v < 8; ++v) {
        const int i = v + 8 * hl;
        gq[i * 32 + l16]      = g0[v];
        gq[i * 32 + 16 + l16] = g1[v];
        gk[i * 32 + l16]      = g2[v];
        gk[i * 32 + 16 + l16] = g3[v];
      }
      // gather Toeplitz bias and commit tile
#pragma unroll
      for (int v = 0; v < 8; ++v) {
        const int i = v + 8 * hl;      // query row in tile
        const int j = l16;             // key col in tile
        const int t = 15 + i - j;      // 0..30
        float val = (s[v] + gq[i * 32 + t] + gk[j * 32 + t]) * 0.125f;
        sc[(rt * 16 + i) * 1024 + r0 + j] = val;
      }
    }
  }
  __syncthreads();

  // ---- pass B: softmax, pack probs bf16 in place --------------------------
  for (int rr = 0; rr < 8; ++rr) {
    const int row = w * 8 + rr;
    float* Sr = sc + row * 1024;
    float mx = -3.0e38f;
    for (int k = lane; k < 1024; k += 32) mx = fmaxf(mx, Sr[k]);
    for (int m = 16; m >= 1; m >>= 1) mx = fmaxf(mx, __shfl_xor(mx, m, 32));
    float sum = 0.f;
    for (int k = lane; k < 1024; k += 32) {
      float e = __expf(Sr[k] - mx);
      Sr[k] = e;
      sum += e;
    }
    for (int m = 16; m >= 1; m >>= 1) sum += __shfl_xor(sum, m, 32);
    const float inv = 1.0f / sum;
    unsigned int* Pu = (unsigned int*)Sr;
    for (int it = 0; it < 16; ++it) {    // pair m reads dwords 2m,2m+1 writes m
      const int m = lane + it * 32;
      v2f f = ((const v2f*)Sr)[m];
      __bf16 x = (__bf16)(f.x * inv), y = (__bf16)(f.y * inv);
      Pu[m] = (unsigned int)__builtin_bit_cast(unsigned short, x) |
              ((unsigned int)__builtin_bit_cast(unsigned short, y) << 16);
    }
  }
  __syncthreads();

  // ---- pass C: ctx = P @ V  (B-frags contiguous thanks to Vt layout) ------
  const __bf16* Pb = (const __bf16*)sc;   // row stride 2048 bf16 (4 KB rows)
  for (int t2 = 0; t2 < 2; ++t2) {
    const int idx = w + t2 * 8;           // 16 tiles: 4 row-tiles x 4 d-tiles
    const int rt = idx >> 2, dt = idx & 3;
    v8f acc = V8F_ZERO;
    for (int ks = 0; ks < 32; ++ks) {
      const int kk = ks * 32 + hl * 16;
      v16bf aP = *(const v16bf*)(Pb + (size_t)(rt * 16 + l16) * 2048 + kk);
      v16bf bV = ldg((const v16bf*)(Vp + (size_t)(dt * 16 + l16) * SEQ + kk));
      acc = WMMA_BF16(aP, bV, acc);
    }
#pragma unroll
    for (int v = 0; v < 8; ++v) {
      const int i = v + 8 * hl;
      const int s = qb * 64 + rt * 16 + i;
      const int d = dt * 16 + l16;
      stg(Ctx + (size_t)(b * SEQ + s) * EMB + h * HD + d, (__bf16)acc[v]);
    }
  }
}

// ---------------------------------------------------------------------------
// Kernel 3: out = LayerNorm(ctx @ Wo + bo + hidden) * gamma + beta
// One workgroup per 16-row tile; wave w owns cols [w*128, w*128+128).
// ---------------------------------------------------------------------------
__global__ void __launch_bounds__(256)
k_out_ln(const __bf16* __restrict__ Ctx, const __bf16* __restrict__ Wot,
         const float* __restrict__ bo, const float* __restrict__ hs,
         const float* __restrict__ gamma, const float* __restrict__ beta,
         float* __restrict__ out) {
  __shared__ float redS[16][8];
  __shared__ float redQ[16][8];
  __shared__ float muA[16];
  __shared__ float rsA[16];

  const int tid  = threadIdx.x;
  const int lane = tid & 31;
  const int w    = tid >> 5;
  const int l16  = lane & 15;
  const int hl   = lane >> 4;
  const int row0 = blockIdx.x * 16;

  v8f acc[8];
#pragma unroll
  for (int q = 0; q < 8; ++q) acc[q] = (v8f)V8F_ZERO;

  for (int ks = 0; ks < 32; ++ks) {
    const int kk = ks * 32 + hl * 16;
    v16bf aC = ldg((const v16bf*)(Ctx + (size_t)(row0 + l16) * EMB + kk));
#pragma unroll
    for (int q = 0; q < 8; ++q) {
      const int n0 = w * 128 + q * 16;
      v16bf bW = ldg((const v16bf*)(Wot + (size_t)(n0 + l16) * 1024 + kk));
      acc[q] = WMMA_BF16(aC, bW, acc[q]);
    }
  }

  // bias + residual, then per-row partial sums
#pragma unroll
  for (int q = 0; q < 8; ++q) {
    const int n = w * 128 + q * 16 + l16;
    const float bias = ldg(bo + n);
#pragma unroll
    for (int v = 0; v < 8; ++v) {
      const int r = row0 + v + 8 * hl;
      acc[q][v] += bias + ldg(hs + (size_t)r * EMB + n);
    }
  }
#pragma unroll
  for (int v = 0; v < 8; ++v) {
    float s = 0.f, s2 = 0.f;
#pragma unroll
    for (int q = 0; q < 8; ++q) {
      float x = acc[q][v];
      s += x;
      s2 += x * x;
    }
    // reduce across the 16 lanes of this half (all same row)
    for (int m = 1; m < 16; m <<= 1) {
      s  += __shfl_xor(s, m, 32);
      s2 += __shfl_xor(s2, m, 32);
    }
    if (l16 == 0) {
      const int i = v + 8 * hl;
      redS[i][w] = s;
      redQ[i][w] = s2;
    }
  }
  __syncthreads();
  if (tid < 16) {
    float S = 0.f, Q = 0.f;
    for (int ww = 0; ww < 8; ++ww) { S += redS[tid][ww]; Q += redQ[tid][ww]; }
    const float mu  = S * (1.0f / EMB);
    const float var = Q * (1.0f / EMB) - mu * mu;
    muA[tid] = mu;
    rsA[tid] = rsqrtf(var + 1e-12f);
  }
  __syncthreads();

#pragma unroll
  for (int q = 0; q < 8; ++q) {
    const int n = w * 128 + q * 16 + l16;
    const float gm = ldg(gamma + n), bt = ldg(beta + n);
#pragma unroll
    for (int v = 0; v < 8; ++v) {
      const int i = v + 8 * hl;
      const int r = row0 + i;
      stg(out + (size_t)r * EMB + n, (acc[q][v] - muA[i]) * rsA[i] * gm + bt);
    }
  }
}

// ---------------------------------------------------------------------------
extern "C" void kernel_launch(void* const* d_in, const int* in_sizes, int n_in,
                              void* d_out, int out_size, void* d_ws, size_t ws_size,
                              hipStream_t stream) {
  const float* hs    = (const float*)d_in[0];
  const float* Wq    = (const float*)d_in[1];
  const float* bq    = (const float*)d_in[2];
  const float* Wk    = (const float*)d_in[3];
  const float* bk    = (const float*)d_in[4];
  const float* Wv    = (const float*)d_in[5];
  const float* bv    = (const float*)d_in[6];
  const float* distE = (const float*)d_in[7];
  const float* Wo    = (const float*)d_in[8];
  const float* bo    = (const float*)d_in[9];
  const float* gam   = (const float*)d_in[10];
  const float* bet   = (const float*)d_in[11];

  char* ws = (char*)d_ws;
  __bf16* Wqt = (__bf16*)(ws + OFF_WQT);
  __bf16* Wkt = (__bf16*)(ws + OFF_WKT);
  __bf16* Wvt = (__bf16*)(ws + OFF_WVT);
  __bf16* Wot = (__bf16*)(ws + OFF_WOT);
  __bf16* Eb  = (__bf16*)(ws + OFF_EB);
  __bf16* Qh  = (__bf16*)(ws + OFF_QH);
  __bf16* Kh  = (__bf16*)(ws + OFF_KH);
  __bf16* Vt  = (__bf16*)(ws + OFF_VT);
  __bf16* Ctx = (__bf16*)(ws + OFF_CTX);

  // precision conversion / layout transforms
  k_transpose_bf16<<<4096, 256, 0, stream>>>(Wq, Wqt);
  k_transpose_bf16<<<4096, 256, 0, stream>>>(Wk, Wkt);
  k_transpose_bf16<<<4096, 256, 0, stream>>>(Wv, Wvt);
  k_transpose_bf16<<<4096, 256, 0, stream>>>(Wo, Wot);
  k_cvt_emb<<<512, 256, 0, stream>>>(distE, Eb);

  // QKV projection
  k_qkv<<<ROWS / 16, 256, 0, stream>>>(hs, Wqt, Wkt, Wvt, bq, bk, bv, Qh, Kh, Vt);

  // attention: 256 KB scores + 32 KB bias staging in LDS (WGP has 320 KB)
  const int attnLds = 64 * 1024 * 4 + 8 * 1024 * 4;   // 294912 B
  (void)hipFuncSetAttribute((const void*)k_attn,
                            hipFuncAttributeMaxDynamicSharedMemorySize, attnLds);
  k_attn<<<BATCH * NH * (SEQ / 64), 256, attnLds, stream>>>(Qh, Kh, Vt, Eb, Ctx);

  // output projection + residual + LayerNorm
  k_out_ln<<<ROWS / 16, 256, 0, stream>>>(Ctx, Wot, bo, hs, gam, bet, (float*)d_out);
}
